// MultiHeadAttention_85444079386691
// MI455X (gfx1250) — matrix operand order repaired and checked by reading, not compiled, not run on hardware
//
#include <hip/hip_runtime.h>

// ---------------------------------------------------------------------------
// MultiHeadAttention forward for MI455X (gfx1250), wave32 + WMMA bf16.
// b=2, n=2048, d_model=1024, heads=16, head_dim=64.
// ---------------------------------------------------------------------------

typedef __attribute__((ext_vector_type(16))) __bf16 v16bf;
typedef __attribute__((ext_vector_type(8)))  float  v8f;

#define D_MODEL 1024
#define N_SEQ   2048
#define N_HEAD  16
#define HD      64
#define BROWS   4096          // b * n
#define LOG2E   1.44269504088896340736f

union U32B2 { unsigned int u; __bf16 b[2]; };

// ---------------------------------------------------------------------------
// WMMA wrapper: D = A(16x32 bf16) * B(32x16 bf16) + C(16x16 f32)
// ---------------------------------------------------------------------------
__device__ __forceinline__ v8f wmma_bf16(v16bf a, v16bf b, v8f c) {
  return __builtin_amdgcn_wmma_f32_16x16x32_bf16(
      /*neg_a=*/false, a, /*neg_b=*/false, b,
      /*c_mod=*/(short)0, c, /*reuse_a=*/false, /*reuse_b=*/false);
}

// A fragment (16x32) from row-major LDS [m][k], leading dim ld (elements).
// ISA: lane m = lane&15; lanes 0-15 hold K {0..7,16..23}, 16-31 {8..15,24..31}.
// ld must be a multiple of 8 so rows are 16B-aligned (enables ds_load_b128).
__device__ __forceinline__ v16bf fragA_row(const __bf16* src, int ld, int lane) {
  const int m  = lane & 15;
  const int kb = (lane >> 4) << 3;
  const __bf16* p = src + m * ld + kb;
  v16bf f;
#pragma unroll
  for (int j = 0; j < 8; ++j) {
    f[j]     = p[j];
    f[j + 8] = p[16 + j];
  }
  return f;
}

// B fragment (32x16) where B^T is stored row-major in LDS as [n][k].
// ISA: lane n = lane&15; lanes 0-15 hold K {0..7,16..23}, 16-31 {8..15,24..31}.
// Two 16B reads per lane -> two ds_load_b128.
__device__ __forceinline__ v16bf fragB_rowT(const __bf16* src, int ld, int lane) {
  const int n  = lane & 15;
  const int kb = (lane >> 4) << 3;
  const __bf16* p = src + n * ld + kb;
  v16bf f;
#pragma unroll
  for (int j = 0; j < 8; ++j) {
    f[j]     = p[j];
    f[j + 8] = p[16 + j];
  }
  return f;
}

// ---------------------------------------------------------------------------
// Kernel 1: projection GEMM  Y = X(f32) @ W(f32) * scale, emitted bf16
// head-split: out[((batch*16+head)*2048 + n)*64 + dd].
// Block = 128 threads (4 waves), 64x64 tile, K-steps of 64.
// ---------------------------------------------------------------------------
__global__ __launch_bounds__(128)
void mha_proj_gemm(const float* __restrict__ X, const float* __restrict__ W,
                   __bf16* __restrict__ Out, float scale) {
  __shared__ __bf16 ldsA [64][72];   // A tile rows [m][k]
  __shared__ __bf16 ldsBt[64][72];   // W tile transposed [n][k]

  const int tid   = threadIdx.x;
  const int lane  = tid & 31;
  const int wave  = tid >> 5;
  const int mBase = blockIdx.y * 64;
  const int nBase = blockIdx.x * 64;
  const int wrow  = (wave >> 1) * 32;
  const int wcol  = (wave & 1) * 32;

  v8f acc[2][2];
#pragma unroll
  for (int i = 0; i < 2; ++i)
#pragma unroll
    for (int j = 0; j < 2; ++j)
#pragma unroll
      for (int r = 0; r < 8; ++r) acc[i][j][r] = 0.0f;

  for (int k0 = 0; k0 < D_MODEL; k0 += 64) {
    __syncthreads();
#pragma unroll
    for (int i = 0; i < 8; ++i) {            // A: 64x64 f32 -> bf16 (float4 loads)
      int idx = tid + i * 128;
      int r = idx >> 4, c4 = (idx & 15) * 4;
      float4 xv = *(const float4*)&X[(size_t)(mBase + r) * D_MODEL + k0 + c4];
      ldsA[r][c4 + 0] = (__bf16)xv.x;
      ldsA[r][c4 + 1] = (__bf16)xv.y;
      ldsA[r][c4 + 2] = (__bf16)xv.z;
      ldsA[r][c4 + 3] = (__bf16)xv.w;
    }
#pragma unroll
    for (int i = 0; i < 8; ++i) {            // B: 64x64 f32 -> bf16, transposed
      int idx = tid + i * 128;
      int r = idx >> 4, c4 = (idx & 15) * 4; // r = k row, c4 = n col
      float4 wv = *(const float4*)&W[(size_t)(k0 + r) * D_MODEL + nBase + c4];
      ldsBt[c4 + 0][r] = (__bf16)wv.x;
      ldsBt[c4 + 1][r] = (__bf16)wv.y;
      ldsBt[c4 + 2][r] = (__bf16)wv.z;
      ldsBt[c4 + 3][r] = (__bf16)wv.w;
    }
    __syncthreads();

#pragma unroll
    for (int ks = 0; ks < 2; ++ks) {
      v16bf a0 = fragA_row (&ldsA [wrow      ][ks * 32], 72, lane);
      v16bf a1 = fragA_row (&ldsA [wrow + 16 ][ks * 32], 72, lane);
      v16bf b0 = fragB_rowT(&ldsBt[wcol      ][ks * 32], 72, lane);
      v16bf b1 = fragB_rowT(&ldsBt[wcol + 16 ][ks * 32], 72, lane);
      acc[0][0] = wmma_bf16(a0, b0, acc[0][0]);
      acc[0][1] = wmma_bf16(a0, b1, acc[0][1]);
      acc[1][0] = wmma_bf16(a1, b0, acc[1][0]);
      acc[1][1] = wmma_bf16(a1, b1, acc[1][1]);
    }
  }

  const int halfsel = lane >> 4;
  const int nlan    = lane & 15;
#pragma unroll
  for (int mi = 0; mi < 2; ++mi)
#pragma unroll
    for (int ni = 0; ni < 2; ++ni)
#pragma unroll
      for (int r = 0; r < 8; ++r) {
        int m = wrow + mi * 16 + r + 8 * halfsel;
        int n = wcol + ni * 16 + nlan;
        int R = mBase + m;                   // 0..4095
        int C = nBase + n;                   // 0..1023
        int batch = R >> 11, nn = R & 2047;
        int head  = C >> 6,  dd = C & 63;
        float v = acc[mi][ni][r] * scale;
        Out[((size_t)((batch * N_HEAD + head) * N_SEQ + nn)) * HD + dd] = (__bf16)v;
      }
}

// ---------------------------------------------------------------------------
// Kernel 2: flash attention. Block = 128 threads (4 waves), one (b,h) and
// 64 query rows; loop over 32 key tiles of 64.
// Q pre-scaled by log2(e)/sqrt(64); mask is multiplicative on logits.
// ---------------------------------------------------------------------------
__global__ __launch_bounds__(128)
void mha_attn(const __bf16* __restrict__ Qh, const __bf16* __restrict__ Kh,
              const __bf16* __restrict__ Vh, const float* __restrict__ mask,
              __bf16* __restrict__ AOut) {
  __shared__ __bf16 ldsQ [64][72];      // [q][dk]
  __shared__ __bf16 ldsK [64][72];      // [key][dk]
  __shared__ __bf16 ldsVt[64][72];      // transposed: [dv][key]
  __shared__ __bf16 ldsP [4][16][72];   // per wave: [m][key]

  const int tid   = threadIdx.x;
  const int lane  = tid & 31;
  const int wave  = tid >> 5;
  const int bh    = blockIdx.y;             // 0..31
  const int batch = bh >> 4;
  const int head  = bh & 15;
  const int q0    = blockIdx.x * 64;

  const __bf16* Qb = Qh + (size_t)bh * N_SEQ * HD;
  const __bf16* Kb = Kh + (size_t)bh * N_SEQ * HD;
  const __bf16* Vb = Vh + (size_t)bh * N_SEQ * HD;
  const unsigned int* K32 = (const unsigned int*)Kb;
  const unsigned int* V32 = (const unsigned int*)Vb;
  const unsigned int* Q32 = (const unsigned int*)Qb;

#pragma unroll
  for (int i = 0; i < 16; ++i) {            // Q tile 64x64 bf16, u32 copies
    int idx = tid + i * 128;
    int r = idx >> 5, cu = idx & 31;
    ((unsigned int*)&ldsQ[r][0])[cu] = Q32[(size_t)(q0 + r) * 32 + cu];
  }
  __syncthreads();

  v16bf aQ[2];
  aQ[0] = fragA_row(&ldsQ[wave * 16][0],  72, lane);
  aQ[1] = fragA_row(&ldsQ[wave * 16][32], 72, lane);

  float m_run[8], l_run[8];
#pragma unroll
  for (int r = 0; r < 8; ++r) { m_run[r] = -3.0e38f; l_run[r] = 0.0f; }
  v8f accO[4];
#pragma unroll
  for (int t = 0; t < 4; ++t)
#pragma unroll
    for (int r = 0; r < 8; ++r) accO[t][r] = 0.0f;

  const int halfsel = lane >> 4;
  const int nlan    = lane & 15;

  for (int kt = 0; kt < N_SEQ / 64; ++kt) {
    const int key0 = kt * 64;
    if (kt + 1 < N_SEQ / 64) {              // prefetch next K/V tiles
      __builtin_prefetch(&Kb[(size_t)(key0 + 64) * HD + tid * 32], 0, 1);
      __builtin_prefetch(&Vb[(size_t)(key0 + 64) * HD + tid * 32], 0, 1);
    }
    __syncthreads();                        // guard ldsK/Vt/P reuse
#pragma unroll
    for (int i = 0; i < 16; ++i) {          // K tile direct, V tile transposed
      int idx = tid + i * 128;
      int r = idx >> 5, cu = idx & 31;
      ((unsigned int*)&ldsK[r][0])[cu] = K32[(size_t)(key0 + r) * 32 + cu];
      U32B2 w; w.u = V32[(size_t)(key0 + r) * 32 + cu];
      ldsVt[cu * 2 + 0][r] = w.b[0];
      ldsVt[cu * 2 + 1][r] = w.b[1];
    }
    __syncthreads();

    // ---- S = Qs * K^T  (16 query rows x 64 keys per wave) ----
    v8f sAcc[4];
#pragma unroll
    for (int ni = 0; ni < 4; ++ni) {
#pragma unroll
      for (int r = 0; r < 8; ++r) sAcc[ni][r] = 0.0f;
      v16bf bK0 = fragB_rowT(&ldsK[ni * 16][0],  72, lane);
      v16bf bK1 = fragB_rowT(&ldsK[ni * 16][32], 72, lane);
      sAcc[ni] = wmma_bf16(aQ[0], bK0, sAcc[ni]);
      sAcc[ni] = wmma_bf16(aQ[1], bK1, sAcc[ni]);
    }

    // ---- multiplicative mask on logits ----
#pragma unroll
    for (int ni = 0; ni < 4; ++ni)
#pragma unroll
      for (int r = 0; r < 8; ++r) {
        int qrow = q0 + wave * 16 + r + 8 * halfsel;
        int key  = key0 + ni * 16 + nlan;
        float mv = mask[((size_t)batch * N_SEQ + qrow) * N_SEQ + key];
        sAcc[ni][r] *= mv;
      }

    // ---- online softmax (rows live in 16-lane halves) ----
#pragma unroll
    for (int r = 0; r < 8; ++r) {
      float tmax = fmaxf(fmaxf(sAcc[0][r], sAcc[1][r]),
                         fmaxf(sAcc[2][r], sAcc[3][r]));
#pragma unroll
      for (int off = 8; off >= 1; off >>= 1)
        tmax = fmaxf(tmax, __shfl_xor(tmax, off, 32));
      float mnew = fmaxf(m_run[r], tmax);
      float corr = __builtin_amdgcn_exp2f(m_run[r] - mnew);
      float rsum = 0.0f;
#pragma unroll
      for (int ni = 0; ni < 4; ++ni) {
        float p = __builtin_amdgcn_exp2f(sAcc[ni][r] - mnew);
        rsum += p;
        ldsP[wave][r + 8 * halfsel][ni * 16 + nlan] = (__bf16)p;
      }
#pragma unroll
      for (int off = 8; off >= 1; off >>= 1)
        rsum += __shfl_xor(rsum, off, 32);
      l_run[r] = l_run[r] * corr + rsum;
      m_run[r] = mnew;
#pragma unroll
      for (int t = 0; t < 4; ++t) accO[t][r] *= corr;
    }
    __syncthreads();                        // P visible in LDS

    // ---- O += P * V ----
#pragma unroll
    for (int ks = 0; ks < 2; ++ks) {
      v16bf aP = fragA_row(&ldsP[wave][0][ks * 32], 72, lane);
#pragma unroll
      for (int t = 0; t < 4; ++t) {
        v16bf bV = fragB_rowT(&ldsVt[t * 16][ks * 32], 72, lane);
        accO[t] = wmma_bf16(aP, bV, accO[t]);
      }
    }
  }

  // ---- normalize and emit bf16 [b, n, h*64] ----
#pragma unroll
  for (int r = 0; r < 8; ++r) {
    float inv = 1.0f / fmaxf(l_run[r], 1e-30f);
    int qrow = q0 + wave * 16 + r + 8 * halfsel;
    size_t rowOff = ((size_t)batch * N_SEQ + qrow) * D_MODEL + head * HD;
#pragma unroll
    for (int t = 0; t < 4; ++t)
      AOut[rowOff + t * 16 + nlan] = (__bf16)(accO[t][r] * inv);
  }
}

// ---------------------------------------------------------------------------
// Kernel 3: output projection  Out = A(bf16) @ Wo(f32) + residual(f32).
// ---------------------------------------------------------------------------
__global__ __launch_bounds__(128)
void mha_out_gemm(const __bf16* __restrict__ A, const float* __restrict__ W,
                  const float* __restrict__ Residual, float* __restrict__ Out) {
  __shared__ __bf16 ldsA [64][72];
  __shared__ __bf16 ldsBt[64][72];

  const int tid   = threadIdx.x;
  const int lane  = tid & 31;
  const int wave  = tid >> 5;
  const int mBase = blockIdx.y * 64;
  const int nBase = blockIdx.x * 64;
  const int wrow  = (wave >> 1) * 32;
  const int wcol  = (wave & 1) * 32;
  const unsigned int* A32 = (const unsigned int*)A;

  v8f acc[2][2];
#pragma unroll
  for (int i = 0; i < 2; ++i)
#pragma unroll
    for (int j = 0; j < 2; ++j)
#pragma unroll
      for (int r = 0; r < 8; ++r) acc[i][j][r] = 0.0f;

  for (int k0 = 0; k0 < D_MODEL; k0 += 64) {
    __syncthreads();
#pragma unroll
    for (int i = 0; i < 16; ++i) {           // A: 64x64 bf16 copy (u32)
      int idx = tid + i * 128;
      int r = idx >> 5, cu = idx & 31;
      ((unsigned int*)&ldsA[r][0])[cu] =
          A32[(size_t)(mBase + r) * (D_MODEL / 2) + k0 / 2 + cu];
    }
#pragma unroll
    for (int i = 0; i < 8; ++i) {            // B: 64x64 f32 -> bf16, transposed
      int idx = tid + i * 128;
      int r = idx >> 4, c4 = (idx & 15) * 4;
      float4 wv = *(const float4*)&W[(size_t)(k0 + r) * D_MODEL + nBase + c4];
      ldsBt[c4 + 0][r] = (__bf16)wv.x;
      ldsBt[c4 + 1][r] = (__bf16)wv.y;
      ldsBt[c4 + 2][r] = (__bf16)wv.z;
      ldsBt[c4 + 3][r] = (__bf16)wv.w;
    }
    __syncthreads();

#pragma unroll
    for (int ks = 0; ks < 2; ++ks) {
      v16bf a0 = fragA_row (&ldsA [wrow      ][ks * 32], 72, lane);
      v16bf a1 = fragA_row (&ldsA [wrow + 16 ][ks * 32], 72, lane);
      v16bf b0 = fragB_rowT(&ldsBt[wcol      ][ks * 32], 72, lane);
      v16bf b1 = fragB_rowT(&ldsBt[wcol + 16 ][ks * 32], 72, lane);
      acc[0][0] = wmma_bf16(a0, b0, acc[0][0]);
      acc[0][1] = wmma_bf16(a0, b1, acc[0][1]);
      acc[1][0] = wmma_bf16(a1, b0, acc[1][0]);
      acc[1][1] = wmma_bf16(a1, b1, acc[1][1]);
    }
  }

  const int halfsel = lane >> 4;
  const int nlan    = lane & 15;
#pragma unroll
  for (int mi = 0; mi < 2; ++mi)
#pragma unroll
    for (int ni = 0; ni < 2; ++ni)
#pragma unroll
      for (int r = 0; r < 8; ++r) {
        int m = wrow + mi * 16 + r + 8 * halfsel;
        int n = wcol + ni * 16 + nlan;
        size_t off = (size_t)(mBase + m) * D_MODEL + nBase + n;
        Out[off] = acc[mi][ni][r] + Residual[off];
      }
}

// ---------------------------------------------------------------------------
// Launcher
// ---------------------------------------------------------------------------
extern "C" void kernel_launch(void* const* d_in, const int* in_sizes, int n_in,
                              void* d_out, int out_size, void* d_ws, size_t ws_size,
                              hipStream_t stream) {
  const float* q    = (const float*)d_in[0];
  const float* k    = (const float*)d_in[1];
  const float* v    = (const float*)d_in[2];
  const float* mask = (const float*)d_in[3];
  const float* wq   = (const float*)d_in[4];
  const float* wk   = (const float*)d_in[5];
  const float* wv   = (const float*)d_in[6];
  const float* wo   = (const float*)d_in[7];
  float* out = (float*)d_out;

  const size_t headElems = (size_t)BROWS * D_MODEL;     // 4M bf16 elements
  __bf16* qh   = (__bf16*)d_ws;
  __bf16* kh   = qh + headElems;
  __bf16* vh   = kh + headElems;
  __bf16* aout = vh + headElems;

  dim3 blk(128);
  dim3 gGemm(D_MODEL / 64, BROWS / 64);                  // (16, 64)
  dim3 gAttn(N_SEQ / 64, 2 * N_HEAD);                    // (32, 32)

  const float qscale = LOG2E / 8.0f;   // log2(e) / sqrt(head_dim)

  mha_proj_gemm<<<gGemm, blk, 0, stream>>>(q, wq, qh, qscale);
  mha_proj_gemm<<<gGemm, blk, 0, stream>>>(k, wk, kh, 1.0f);
  mha_proj_gemm<<<gGemm, blk, 0, stream>>>(v, wv, vh, 1.0f);
  mha_attn<<<gAttn, blk, 0, stream>>>(qh, kh, vh, mask, aout);
  mha_out_gemm<<<gGemm, blk, 0, stream>>>(aout, wo, q, out);
}
